// AutopoieticMoELayer_62783831933690
// MI455X (gfx1250) — compile-verified, matrix operand order checked by reading, not run on hardware
//
#include <hip/hip_runtime.h>
#include <cstdint>
#include <cstddef>

// ---------------------------------------------------------------------------
// MoE layer for MI455X (gfx1250, wave32).
//   T=8192 tokens, D=2048, H=2048, E=16, top_k=2.
// zero-init -> router (top-2, bucket scatter) -> grouped bf16-WMMA GEMM with
// weighted atomic combine (exactly 2 commutative f32 adds/element => bitwise
// deterministic).  Double-buffered LDS pipeline, 1 barrier / K-step,
// 64 v_wmma_f32_16x16x32_bf16 per block per barrier.
// ---------------------------------------------------------------------------

typedef __attribute__((ext_vector_type(16))) __bf16 v16bf;
typedef __attribute__((ext_vector_type(8)))  __bf16 v8bf;
typedef __attribute__((ext_vector_type(4)))  __bf16 v4bf;
typedef __attribute__((ext_vector_type(8)))  float  v8f;

constexpr int T_TOK = 8192;   // B*S
constexpr int DDIM  = 2048;
constexpr int HDIM  = 2048;
constexpr int NEXP  = 16;

// ws layout: [0,64) counts | [256, +E*T*4) bucket idx | next E*T*4 bucket w
constexpr size_t WS_IDX_OFF = 256;
constexpr size_t WS_W_OFF   = WS_IDX_OFF + (size_t)NEXP * T_TOK * 4;

// ---------------------------------------------------------------------------
// Router: one wave per token; butterfly-reduced logits; top-2 on logits
// (softmax monotonic; renormalized weights = sigmoid of logit gap).
// ---------------------------------------------------------------------------
__global__ __launch_bounds__(256) void moe_router(
    const float* __restrict__ x, const float* __restrict__ Wr,
    const float* __restrict__ br,
    int* __restrict__ counts, int* __restrict__ bidx, float* __restrict__ bw)
{
  const int lane = threadIdx.x & 31;
  const int wid  = threadIdx.x >> 5;
  const int t    = blockIdx.x * 8 + wid;

  float acc[NEXP];
#pragma unroll
  for (int e = 0; e < NEXP; ++e) acc[e] = 0.0f;

  const float4* xv = reinterpret_cast<const float4*>(x + (size_t)t * DDIM);
  const float4* wv = reinterpret_cast<const float4*>(Wr);
#pragma unroll 4
  for (int i = 0; i < DDIM / 128; ++i) {
    float4 xf = xv[i * 32 + lane];
#pragma unroll
    for (int e = 0; e < NEXP; ++e) {
      float4 wf = wv[e * (DDIM / 4) + i * 32 + lane];
      acc[e] += xf.x * wf.x + xf.y * wf.y + xf.z * wf.z + xf.w * wf.w;
    }
  }
#pragma unroll
  for (int e = 0; e < NEXP; ++e) {
#pragma unroll
    for (int off = 16; off > 0; off >>= 1)
      acc[e] += __shfl_xor(acc[e], off, 32);
    acc[e] += br[e];
  }

  float l0 = -1e30f, l1 = -1e30f;
  int   e0 = 0,      e1 = 0;
#pragma unroll
  for (int e = 0; e < NEXP; ++e) {
    float v = acc[e];
    if (v > l0)      { l1 = l0; e1 = e0; l0 = v; e0 = e; }
    else if (v > l1) { l1 = v;  e1 = e; }
  }
  float r  = __expf(l1 - l0);
  float w0 = 1.0f / (1.0f + r);
  float w1 = 1.0f - w0;

  if (lane == 0) {
    int p0 = atomicAdd(&counts[e0], 1);
    bidx[e0 * T_TOK + p0] = t;  bw[e0 * T_TOK + p0] = w0;
    int p1 = atomicAdd(&counts[e1], 1);
    bidx[e1 * T_TOK + p1] = t;  bw[e1 * T_TOK + p1] = w1;
  }
}

// ---------------------------------------------------------------------------
// Grouped expert GEMM.  Grid (T/128, H/128, E); block 256 thr = 8 waves,
// waves 4(M) x 2(N); each wave: 32x64 patch = 2x4 WMMA frags (8 wmma/K-step).
// Double-buffered LDS, software pipeline, one workgroup barrier per K-step.
// ---------------------------------------------------------------------------
#define LDA 40   // padded bf16 row stride: 80B keeps 16B alignment, spreads banks

__global__ __launch_bounds__(256) void moe_expert_gemm(
    const float* __restrict__ x, const float* __restrict__ We,
    const float* __restrict__ be,
    const int* __restrict__ counts, const int* __restrict__ bidx,
    const float* __restrict__ bw, float* __restrict__ out)
{
  const int e   = blockIdx.z;
  const int cnt = counts[e];
  const int m0  = blockIdx.x * 128;
  if (m0 >= cnt) return;
  const int n0  = blockIdx.y * 128;

  const int tid   = threadIdx.x;
  const int lane  = tid & 31;
  const int wid   = tid >> 5;
  const int waveM = wid & 3;   // 0..3 -> 32-row slice
  const int waveN = wid >> 2;  // 0..1 -> 64-col slice

  __shared__ __bf16 As[2][128 * LDA];
  __shared__ __bf16 Bs[2][128 * LDA];
  __shared__ int    stok[128];
  __shared__ float  swt[128];

  // block-wide gather metadata (tokens + routing weights), staged once
  if (tid < 128) {
    const int gr = m0 + tid;
    stok[tid] = (gr < cnt) ? bidx[e * T_TOK + gr] : 0;
    swt[tid]  = (gr < cnt) ? bw[e * T_TOK + gr]   : 0.0f;
  }
  __syncthreads();

  // staging map: 2 threads per row, 16 contiguous floats each
  const int srow  = tid >> 1;          // 0..127
  const int shalf = (tid & 1) * 16;    // 0 or 16
  const float* aptr = x  + (size_t)stok[srow] * DDIM + shalf;
  const float* bptr = We + ((size_t)e * HDIM + (n0 + srow)) * DDIM + shalf;
  const int     sst = srow * LDA + shalf;

  float4 ra[4], rb[4];
  auto load_tiles = [&](int k0) {
    const float4* ap = reinterpret_cast<const float4*>(aptr + k0);
    const float4* bp = reinterpret_cast<const float4*>(bptr + k0);
#pragma unroll
    for (int j = 0; j < 4; ++j) { ra[j] = ap[j]; rb[j] = bp[j]; }
  };
  auto store_tiles = [&](int s) {
#pragma unroll
    for (int j = 0; j < 4; ++j) {
      v4bf ca; ca.x = (__bf16)ra[j].x; ca.y = (__bf16)ra[j].y;
               ca.z = (__bf16)ra[j].z; ca.w = (__bf16)ra[j].w;
      *reinterpret_cast<v4bf*>(&As[s][sst + j * 4]) = ca;
      v4bf cb; cb.x = (__bf16)rb[j].x; cb.y = (__bf16)rb[j].y;
               cb.z = (__bf16)rb[j].z; cb.w = (__bf16)rb[j].w;
      *reinterpret_cast<v4bf*>(&Bs[s][sst + j * 4]) = cb;
    }
  };

  v8f acc[2][4];
#pragma unroll
  for (int i = 0; i < 2; ++i)
#pragma unroll
    for (int j = 0; j < 4; ++j)
#pragma unroll
      for (int k = 0; k < 8; ++k) acc[i][j][k] = 0.0f;

  // WMMA 16-bit A/B fragment addressing (ISA 7.12.2):
  //   lane 0-15 : row/col = lane,    k 0..7  and 16..23
  //   lane16-31 : row/col = lane-16, k 8..15 and 24..31
  const int frow  = lane & 15;
  const int klo   = (lane >> 4) * 8;
  const int aoff0 = (waveM * 32 + frow) * LDA + klo;
  const int boff0 = (waveN * 64 + frow) * LDA + klo;

  constexpr int NIT = DDIM / 32;   // 64 K-steps

  // pipeline prologue: tile0 -> buf0, tile1 -> regs
  load_tiles(0);
  store_tiles(0);
  load_tiles(32);
  __syncthreads();

  for (int it = 0; it < NIT; ++it) {
    const int cur = it & 1;
    if (it + 1 < NIT) store_tiles(cur ^ 1);      // regs -> other buffer
    if (it + 2 < NIT) load_tiles((it + 2) * 32); // prefetch next-next tile

    v16bf af[2], bfr[4];
#pragma unroll
    for (int i = 0; i < 2; ++i) {
      v8bf lo = *reinterpret_cast<const v8bf*>(&As[cur][aoff0 + i * 16 * LDA]);
      v8bf hi = *reinterpret_cast<const v8bf*>(&As[cur][aoff0 + i * 16 * LDA + 16]);
      af[i] = __builtin_shufflevector(lo, hi, 0,1,2,3,4,5,6,7,8,9,10,11,12,13,14,15);
    }
#pragma unroll
    for (int j = 0; j < 4; ++j) {
      v8bf lo = *reinterpret_cast<const v8bf*>(&Bs[cur][boff0 + j * 16 * LDA]);
      v8bf hi = *reinterpret_cast<const v8bf*>(&Bs[cur][boff0 + j * 16 * LDA + 16]);
      bfr[j] = __builtin_shufflevector(lo, hi, 0,1,2,3,4,5,6,7,8,9,10,11,12,13,14,15);
    }
#pragma unroll
    for (int i = 0; i < 2; ++i)
#pragma unroll
      for (int j = 0; j < 4; ++j)
        acc[i][j] = __builtin_amdgcn_wmma_f32_16x16x32_bf16(
            false, af[i], false, bfr[j], (short)0, acc[i][j], false, false);

    __syncthreads();   // single barrier per K-step (double buffer)
  }

  // Epilogue.  C frag layout: lane 0-15 -> N=lane, M=j; lane 16-31 -> N=lane-16, M=8+j.
  const int crow = (lane >> 4) * 8;
  const int ccol = lane & 15;
  float bias[4];
#pragma unroll
  for (int j = 0; j < 4; ++j)
    bias[j] = be[e * HDIM + n0 + waveN * 64 + j * 16 + ccol];

#pragma unroll
  for (int i = 0; i < 2; ++i) {
#pragma unroll
    for (int rr = 0; rr < 8; ++rr) {
      const int rloc = waveM * 32 + i * 16 + crow + rr;
      if (m0 + rloc < cnt) {
        const int   tok = stok[rloc];
        const float w   = swt[rloc];
        float* obase = out + (size_t)tok * HDIM + n0 + waveN * 64 + ccol;
#pragma unroll
        for (int j = 0; j < 4; ++j)
          unsafeAtomicAdd(obase + j * 16, w * (acc[i][j][rr] + bias[j]));
      }
    }
  }
}

// ---------------------------------------------------------------------------
extern "C" void kernel_launch(void* const* d_in, const int* in_sizes, int n_in,
                              void* d_out, int out_size, void* d_ws, size_t ws_size,
                              hipStream_t stream) {
  const float* x  = (const float*)d_in[0];
  const float* Wr = (const float*)d_in[1];
  const float* br = (const float*)d_in[2];
  const float* We = (const float*)d_in[3];
  const float* be = (const float*)d_in[4];
  // d_in[5] = top_k (==2, baked in)

  int*   counts = (int*)d_ws;
  int*   bidx   = (int*)((char*)d_ws + WS_IDX_OFF);
  float* bw     = (float*)((char*)d_ws + WS_W_OFF);
  float* out    = (float*)d_out;

  hipMemsetAsync(d_out, 0, (size_t)out_size * sizeof(float), stream);
  hipMemsetAsync(d_ws, 0, WS_IDX_OFF, stream);   // zero expert counters

  moe_router<<<T_TOK / 8, 256, 0, stream>>>(x, Wr, br, counts, bidx, bw);

  dim3 grid(T_TOK / 128, HDIM / 128, NEXP);
  moe_expert_gemm<<<grid, 256, 0, stream>>>(x, We, be, counts, bidx, bw, out);
}
